// SharedQueryMultiHeadAttention_52381421142744
// MI455X (gfx1250) — matrix operand order repaired and checked by reading, not compiled, not run on hardware
//
#include <hip/hip_runtime.h>
#include <hip/hip_bf16.h>

#define EMBED 2048
#define NHEAD 16
#define HDIM  128
#define BATCH 4
#define SEQ   2048

typedef __attribute__((ext_vector_type(16))) __bf16 v16bf;
typedef __attribute__((ext_vector_type(8)))  __bf16 v8bf;
typedef __attribute__((ext_vector_type(8)))  float  v8f;

// ---------------------------------------------------------------------------
// CDNA5 async global->LDS copy (ASYNCcnt path), with sync fallback
// ---------------------------------------------------------------------------
#if defined(__gfx1250__) && __has_builtin(__builtin_amdgcn_global_load_async_to_lds_b128)
#define HAVE_ASYNC_LDS 1
#endif

typedef int v4i_vs __attribute__((vector_size(4 * sizeof(int))));
typedef __attribute__((address_space(1))) v4i_vs as1_v4i;
typedef __attribute__((address_space(3))) v4i_vs as3_v4i;

static __device__ __forceinline__ void async_copy16(const void* g, void* l) {
#if defined(HAVE_ASYNC_LDS)
  __builtin_amdgcn_global_load_async_to_lds_b128((as1_v4i*)g, (as3_v4i*)l, 0, 0);
#else
  *(v8bf*)l = *(const v8bf*)g;
#endif
}

static __device__ __forceinline__ void wait_async_lds() {
#if defined(HAVE_ASYNC_LDS)
#if __has_builtin(__builtin_amdgcn_s_wait_asynccnt)
  __builtin_amdgcn_s_wait_asynccnt(0);
#else
  asm volatile("s_wait_asynccnt 0x0" ::: "memory");
#endif
#endif
}

// ---------------------------------------------------------------------------
// WMMA fragment helpers (CDNA5 ISA 7.12.2 layouts, wave32)
// ---------------------------------------------------------------------------

// Build an A-fragment from a per-lane pointer to its 16 contiguous K elements
// (K split at lane 16: lane half h holds k = 16h + s).
static __device__ __forceinline__ v16bf frag_a(const __bf16* p) {
  v8bf lo = *(const v8bf*)(p);
  v8bf hi = *(const v8bf*)(p + 8);
  return __builtin_shufflevector(lo, hi, 0, 1, 2, 3, 4, 5, 6, 7,
                                         8, 9, 10, 11, 12, 13, 14, 15);
}

// A-matrix 16x32 bf16 from row-major source, leading dim ldk (elements).
static __device__ __forceinline__ v16bf load_a16x32(const __bf16* tile, int ldk, int lane) {
  const int m  = lane & 15;
  const int kh = lane >> 4;
  return frag_a(tile + (size_t)m * ldk + kh * 16);
}

// B-matrix 32x16 tile: lane l -> column N = l&15, contiguous K seg (l>>4)*16.
static __device__ __forceinline__ v16bf load_b32x16_T(const __bf16* wtile, int ldk, int lane) {
  const int n  = lane & 15;
  const int ks = (lane >> 4) * 16;
  return *(const v16bf*)(wtile + (size_t)n * ldk + ks);
}

static __device__ __forceinline__ v8f wmma_bf16(v16bf a, v16bf b, v8f c) {
  return __builtin_amdgcn_wmma_f32_16x16x32_bf16(false, a, false, b, (short)0, c,
                                                 false, false);
}

// ---------------------------------------------------------------------------
// fp32 -> bf16 conversion (grid-stride)
// ---------------------------------------------------------------------------
__global__ void __launch_bounds__(256)
cvt_f32_bf16_kernel(const float* __restrict__ src, __bf16* __restrict__ dst, size_t n) {
  size_t i = (size_t)blockIdx.x * blockDim.x + threadIdx.x;
  size_t stride = (size_t)gridDim.x * blockDim.x;
  for (; i < n; i += stride) dst[i] = (__bf16)src[i];
}

// ---------------------------------------------------------------------------
// GEMM: Y[M,N] = X[M,K] * W[N,K]^T + bias[N]
// block = 128 threads (4 waves). Block tile 64x64, K-step 64, double-buffered
// LDS staging via async global->LDS copies. Wave w owns rows [w*16,+16);
// B panel shared by all 4 waves. Per k-step: load ALL fragments (one ds
// clause + single wait), then 8 back-to-back WMMAs on 4 accumulators.
// waves_per_eu(1) relaxes the VGPR budget so all 8 B-fragments stay live.
// storeVT!=0 writes bf16 output transposed as [batch][N][SEQ] (V projection).
// ---------------------------------------------------------------------------
#define KSTEP 64
#define LPAD  72  // 64 + 8 elements pad: stride 36 banks (gcd 4) -> conflict-free

__global__ void __launch_bounds__(128)
__attribute__((amdgpu_waves_per_eu(1)))
gemm_xwT_bias_kernel(const __bf16* __restrict__ X, const __bf16* __restrict__ W,
                     const float* __restrict__ bias,
                     float* __restrict__ Yf, __bf16* __restrict__ Yb,
                     int M, int N, int K, int storeVT) {
  __shared__ __bf16 As[2][64 * LPAD];
  __shared__ __bf16 Bs[2][64 * LPAD];

  const int tid  = threadIdx.x;
  const int lane = tid & 31;
  const int wave = tid >> 5;
  const int m0b = blockIdx.y * 64;
  const int n0b = blockIdx.x * 64;

  // staging: 64 rows x 64B (8 of each 16 el) per matrix = 256 16B transfers; 2 per thread.
  const int r0 = tid >> 2;               // rows 0..31
  const int r1 = r0 + 32;                // rows 32..63
  const int sg = (tid & 3) * 16;         // 16-element segment in the 64-el row

  // per-lane fragment coordinates
  const int m  = lane & 15;
  const int kh = lane >> 4;
  const int ks = kh * 16;
  const int aoff = (wave * 16 + m) * LPAD + kh * 16;  // A frag base (s=0)
  const int boff = m * LPAD + ks;                    // B frag base (n == m bits)

  v8f acc[4] = {v8f{}, v8f{}, v8f{}, v8f{}};

  int buf = 0;
  {
    async_copy16(X + (size_t)(m0b + r0) * K + sg, &As[0][r0 * LPAD + sg]);
    async_copy16(X + (size_t)(m0b + r1) * K + sg, &As[0][r1 * LPAD + sg]);
    async_copy16(W + (size_t)(n0b + r0) * K + sg, &Bs[0][r0 * LPAD + sg]);
    async_copy16(W + (size_t)(n0b + r1) * K + sg, &Bs[0][r1 * LPAD + sg]);
  }

  for (int k0 = 0; k0 < K; k0 += KSTEP) {
    wait_async_lds();
    __syncthreads();                      // current buffer filled & visible
    if (k0 + KSTEP < K) {
      const int nb = buf ^ 1;
      const int kn = k0 + KSTEP;
      async_copy16(X + (size_t)(m0b + r0) * K + kn + sg, &As[nb][r0 * LPAD + sg]);
      async_copy16(X + (size_t)(m0b + r1) * K + kn + sg, &As[nb][r1 * LPAD + sg]);
      async_copy16(W + (size_t)(n0b + r0) * K + kn + sg, &Bs[nb][r0 * LPAD + sg]);
      async_copy16(W + (size_t)(n0b + r1) * K + kn + sg, &Bs[nb][r1 * LPAD + sg]);
    }

    // ---- load ALL fragments for this 64-deep k-step, then MMA burst -------
    const __bf16* aBase = &As[buf][aoff];
    const __bf16* bBase = &Bs[buf][boff];
    v16bf a0 = frag_a(aBase);        // k = k0 + [0,32)
    v16bf a1 = frag_a(aBase + 32);   // k = k0 + [32,64)
    v16bf bb0[4], bb1[4];
#pragma unroll
    for (int t = 0; t < 4; ++t) {
      bb0[t] = *(const v16bf*)(bBase + t * 16 * LPAD);
      bb1[t] = *(const v16bf*)(bBase + t * 16 * LPAD + 32);
    }
#pragma unroll
    for (int t = 0; t < 4; ++t) acc[t] = wmma_bf16(a0, bb0[t], acc[t]);
#pragma unroll
    for (int t = 0; t < 4; ++t) acc[t] = wmma_bf16(a1, bb1[t], acc[t]);

    __syncthreads();                      // everyone done reading buf
    buf ^= 1;
  }

  // C/D layout: lane l -> N = l&15, M = (l>>4)*8 + r.
  const int n = lane & 15;
  const int mb = (lane >> 4) * 8;
  const int m0 = m0b + wave * 16;
#pragma unroll
  for (int t = 0; t < 4; ++t) {
    const int nc = n0b + 16 * t + n;
    const float bv = bias[nc];
    if (storeVT) {
#pragma unroll
      for (int r = 0; r < 8; ++r) {
        const int row = m0 + mb + r;
        const int bb = row >> 11;          // row / SEQ
        const int ss = row & (SEQ - 1);
        Yb[((size_t)bb * N + nc) * SEQ + ss] = (__bf16)(acc[t][r] + bv);
      }
    } else if (Yb) {
#pragma unroll
      for (int r = 0; r < 8; ++r)
        Yb[(size_t)(m0 + mb + r) * N + nc] = (__bf16)(acc[t][r] + bv);
    } else {
#pragma unroll
      for (int r = 0; r < 8; ++r)
        Yf[(size_t)(m0 + mb + r) * N + nc] = acc[t][r] + bv;
    }
  }
}

// ---------------------------------------------------------------------------
// MQA attention core. grid = (S/16, H, B), block = 256 (8 waves).
// Dynamic LDS: fp32 scores [16][SC_STRIDE], reduction scratch, and a bf16
// probability buffer [16][PB_STRIDE] laid out for direct A-fragment loads.
//   phase 1: wave w computes scores for keys [w*256, +256) via WMMA
//   phase 2: workgroup softmax; normalized attn -> d_out; bf16 probs -> LDS
//   phase 3: wave w computes ctx hd-tile [w*16, +16) = P @ V via WMMA,
//            with V pre-transposed ([b][hd][key]) so B-frags are contiguous.
// ---------------------------------------------------------------------------
#define SC_STRIDE 2064  // fp32 scores row stride (2048 + 16 pad)
#define PB_STRIDE 2056  // bf16 probs row stride  (2048 + 8 pad, 16B aligned)

__global__ void __launch_bounds__(256)
__attribute__((amdgpu_waves_per_eu(1)))
mqa_attention_kernel(const __bf16* __restrict__ Qp,   // [B*S, EMBED]
                     const __bf16* __restrict__ Kp,   // [B*S, HDIM]
                     const __bf16* __restrict__ Vt,   // [B, HDIM, SEQ]
                     float* __restrict__ attn,        // [B,H,S,S]
                     __bf16* __restrict__ ctx) {      // [B*S, EMBED]
  extern __shared__ float smem[];
  float*  sc   = smem;                        // 16 * SC_STRIDE fp32
  float*  red  = smem + 16 * SC_STRIDE;       // 256 fp32
  float*  rowv = red + 256;                   // 16 fp32
  __bf16* pb   = (__bf16*)(rowv + 16);        // 16 * PB_STRIDE bf16 (16B aligned)

  const int q0 = blockIdx.x * 16;
  const int h  = blockIdx.y;
  const int b  = blockIdx.z;
  const int tid  = threadIdx.x;
  const int lane = tid & 31;
  const int wave = tid >> 5;

  const float scale = 0.08838834764831845f;  // 128^-0.5

  // ---- phase 1: scores = scale * Q K^T ------------------------------------
  {
    v16bf qf[4];
    const __bf16* qtile = Qp + ((size_t)b * SEQ + q0) * EMBED + h * HDIM;
#pragma unroll
    for (int c = 0; c < 4; ++c)
      qf[c] = load_a16x32(qtile + c * 32, EMBED, lane);

    const __bf16* kbase = Kp + (size_t)b * SEQ * HDIM;
    const int n = lane & 15;
    const int mb = (lane >> 4) * 8;
#pragma unroll 2
    for (int t = 0; t < 16; ++t) {
      const int kt = wave * 256 + t * 16;  // key tile base
      // load all 4 K fragments, then chain 4 WMMAs
      v16bf kb[4];
#pragma unroll
      for (int cc = 0; cc < 4; ++cc)
        kb[cc] = load_b32x16_T(kbase + (size_t)kt * HDIM + cc * 32, HDIM, lane);
      v8f c = v8f{};
#pragma unroll
      for (int cc = 0; cc < 4; ++cc) c = wmma_bf16(qf[cc], kb[cc], c);
#pragma unroll
      for (int r = 0; r < 8; ++r)
        sc[(mb + r) * SC_STRIDE + kt + n] = c[r] * scale;
    }
  }
  __syncthreads();

  // ---- phase 2: softmax over 2048 keys; write attn + bf16 probs ------------
  {
    const int row = tid >> 4;   // 0..15 query row
    const int seg = tid & 15;   // 0..15 segment of 128 keys
    float*  rp = sc + row * SC_STRIDE + seg * 128;
    __bf16* pp = pb + row * PB_STRIDE + seg * 128;

    float mx = -3.402823466e38f;
#pragma unroll 4
    for (int i = 0; i < 128; ++i) mx = fmaxf(mx, rp[i]);
    red[tid] = mx;
    __syncthreads();
    if (seg == 0) {
      float v = red[row * 16];
#pragma unroll
      for (int j = 1; j < 16; ++j) v = fmaxf(v, red[row * 16 + j]);
      rowv[row] = v;
    }
    __syncthreads();
    const float rm = rowv[row];
    float s = 0.f;
#pragma unroll 4
    for (int i = 0; i < 128; ++i) {
      float e = __expf(rp[i] - rm);
      rp[i] = e;
      s += e;
    }
    red[tid] = s;
    __syncthreads();
    if (seg == 0) {
      float v = 0.f;
#pragma unroll
      for (int j = 0; j < 16; ++j) v += red[row * 16 + j];
      rowv[row] = 1.0f / v;
    }
    __syncthreads();
    const float inv = rowv[row];
    float* ap = attn + (((size_t)b * NHEAD + h) * SEQ + q0 + row) * SEQ + seg * 128;
#pragma unroll 4
    for (int i = 0; i < 128; ++i) {
      float p = rp[i] * inv;
      ap[i] = p;
      pp[i] = (__bf16)p;
    }
  }
  __syncthreads();

  // ---- phase 3: ctx = P @ V ; wave w owns head-dim tile [w*16, +16) --------
  {
    const int n0h = wave * 16;
    const int n  = lane & 15;
    const int kh = lane >> 4;
    const int ks = kh * 16;
    v8f acc = v8f{};
    const __bf16* vtrow = Vt + ((size_t)b * HDIM + n0h + n) * SEQ;  // contiguous keys
    const __bf16* prow  = pb + (lane & 15) * PB_STRIDE + kh * 16;
#pragma unroll 4
    for (int kk = 0; kk < SEQ; kk += 32) {
      v16bf a  = frag_a(prow + kk);
      v16bf bf = *(const v16bf*)(vtrow + kk + ks);
      acc = wmma_bf16(a, bf, acc);
    }
    const int mb = (lane >> 4) * 8;
#pragma unroll
    for (int r = 0; r < 8; ++r)
      ctx[((size_t)b * SEQ + q0 + mb + r) * EMBED + h * HDIM + n0h + n] = (__bf16)acc[r];
  }
}

// ---------------------------------------------------------------------------
// Launcher
// ---------------------------------------------------------------------------
extern "C" void kernel_launch(void* const* d_in, const int* in_sizes, int n_in,
                              void* d_out, int out_size, void* d_ws, size_t ws_size,
                              hipStream_t stream) {
  (void)in_sizes; (void)n_in; (void)out_size; (void)ws_size;

  const float* q_f  = (const float*)d_in[0];
  const float* k_f  = (const float*)d_in[1];
  const float* v_f  = (const float*)d_in[2];
  const float* Wq_f = (const float*)d_in[3];
  const float* bq   = (const float*)d_in[4];
  const float* Wk_f = (const float*)d_in[5];
  const float* bk   = (const float*)d_in[6];
  const float* Wv_f = (const float*)d_in[7];
  const float* bv   = (const float*)d_in[8];
  const float* Wo_f = (const float*)d_in[9];
  const float* bo   = (const float*)d_in[10];

  const size_t BSE = (size_t)BATCH * SEQ * EMBED;  // 16,777,216
  const size_t BSH = (size_t)BATCH * SEQ * HDIM;   //  1,048,576
  const size_t EE  = (size_t)EMBED * EMBED;        //  4,194,304
  const size_t HE  = (size_t)HDIM * EMBED;         //    262,144

  // workspace layout (bf16 buffers), 256B-aligned
  char* ws = (char*)d_ws;
  size_t off = 0;
  auto carve = [&](size_t elems) {
    void* p = ws + off;
    off += (elems * sizeof(__bf16) + 255) & ~(size_t)255;
    return (__bf16*)p;
  };
  __bf16* q_b    = carve(BSE);
  __bf16* k_b    = carve(BSE);
  __bf16* v_b    = carve(BSE);
  __bf16* Wq_b   = carve(EE);
  __bf16* Wk_b   = carve(HE);
  __bf16* Wv_b   = carve(HE);
  __bf16* Wo_b   = carve(EE);
  __bf16* qproj  = carve(BSE);
  __bf16* kproj  = carve(BSH);
  __bf16* vprojT = carve(BSH);   // [B][HDIM][SEQ]
  __bf16* ctx    = carve(BSE);

  // 1) convert fp32 inputs/weights to bf16
  {
    dim3 blk(256), grd(4096);
    cvt_f32_bf16_kernel<<<grd, blk, 0, stream>>>(q_f, q_b, BSE);
    cvt_f32_bf16_kernel<<<grd, blk, 0, stream>>>(k_f, k_b, BSE);
    cvt_f32_bf16_kernel<<<grd, blk, 0, stream>>>(v_f, v_b, BSE);
    cvt_f32_bf16_kernel<<<grd, blk, 0, stream>>>(Wq_f, Wq_b, EE);
    cvt_f32_bf16_kernel<<<grd, blk, 0, stream>>>(Wk_f, Wk_b, HE);
    cvt_f32_bf16_kernel<<<grd, blk, 0, stream>>>(Wv_f, Wv_b, HE);
    cvt_f32_bf16_kernel<<<grd, blk, 0, stream>>>(Wo_f, Wo_b, EE);
  }

  const int M = BATCH * SEQ;  // 8192

  // 2) projections (bf16 out, f32 accum); V stored transposed per batch
  gemm_xwT_bias_kernel<<<dim3(EMBED / 64, M / 64), 128, 0, stream>>>(
      q_b, Wq_b, bq, nullptr, qproj, M, EMBED, EMBED, 0);
  gemm_xwT_bias_kernel<<<dim3(HDIM / 64, M / 64), 128, 0, stream>>>(
      k_b, Wk_b, bk, nullptr, kproj, M, HDIM, EMBED, 0);
  gemm_xwT_bias_kernel<<<dim3(HDIM / 64, M / 64), 128, 0, stream>>>(
      v_b, Wv_b, bv, nullptr, vprojT, M, HDIM, EMBED, 1);

  // 3) attention: scores + softmax + attn store + ctx
  float* out_f  = (float*)d_out;                 // [B,S,E]
  float* attn_f = out_f + BSE;                   // [B,H,S,S]
  {
    size_t lds = (16 * SC_STRIDE + 256 + 16) * sizeof(float) +
                 (size_t)16 * PB_STRIDE * sizeof(__bf16);      // ~194.3 KB
    mqa_attention_kernel<<<dim3(SEQ / 16, NHEAD, BATCH), 256, lds, stream>>>(
        qproj, kproj, vprojT, attn_f, ctx);
  }

  // 4) output projection (fp32 out)
  gemm_xwT_bias_kernel<<<dim3(EMBED / 64, M / 64), 128, 0, stream>>>(
      ctx, Wo_b, bo, out_f, nullptr, M, EMBED, EMBED, 0);
}
